// MultiHeadSelfAttentionWithRelBias_41601053229171
// MI455X (gfx1250) — compile-verified, matrix operand order checked by reading, not run on hardware
//
#include <hip/hip_runtime.h>
#include <stdint.h>

// ---- problem constants (from reference) ----
#define DIMN   1024
#define NHEAD  16
#define HD     64
#define LSEQ   2048
#define BATCH  2
#define MROWS  (BATCH*LSEQ)   // 4096
#define NQKV   (3*DIMN)       // 3072
static constexpr float SCALE = 0.125f;  // HD^-0.5

typedef __attribute__((ext_vector_type(16))) __bf16 v16bf;
typedef __attribute__((ext_vector_type(8)))  __bf16 v8bf;
typedef __attribute__((ext_vector_type(8)))  float  v8f;
typedef __attribute__((ext_vector_type(4)))  unsigned int v4u;
typedef __attribute__((ext_vector_type(8)))  int v8i;
typedef __attribute__((ext_vector_type(4)))  int v4i;

// ---------------- WMMA helpers ----------------
__device__ __forceinline__ v8f wmma_bf16(v16bf a, v16bf b, v8f c) {
  return __builtin_amdgcn_wmma_f32_16x16x32_bf16(false, a, false, b, (short)0, c,
                                                 false, false);
}

// Fragment from a row-major panel: lane l -> row (l&15) (folded into p by
// caller); element 0..7 = K[hi*8..+7], 8..15 = K[16+hi*8..+7]. Two 16B loads.
__device__ __forceinline__ v16bf frag16(const __bf16* p, int hi) {
  v8bf lo = *(const v8bf*)(p + hi * 8);
  v8bf hh = *(const v8bf*)(p + 16 + hi * 8);
  return __builtin_shufflevector(lo, hh, 0,1,2,3,4,5,6,7,8,9,10,11,12,13,14,15);
}

// Global-memory fragment (row stride ld elements, chunk base k0).
__device__ __forceinline__ v16bf gfrag(const __bf16* __restrict__ base,
                                       int ld, int row, int k0, int hi) {
  return frag16(base + (size_t)row * (size_t)ld + k0, hi);
}

// ---------------- Tensor Data Mover ----------------
// Low 32 bits of a generic LDS pointer == LDS byte offset (flat aperture map).
__device__ __forceinline__ unsigned lds_off(const void* p) {
  return (unsigned)(uintptr_t)p;
}

// Issue a 2-D TDM tile load of bf16 data: tile_w elems along the contiguous
// dim0, tile_h rows, source row stride `stride` elems. D# packing per
// ISA 8.3/8.4: group0 = {count=1 | lds_addr | global_addr | type=2},
// group1 = {data_size=2B, tensor_dim0/1, tile_dim0/1, tensor_dim0_stride}.
__device__ __forceinline__ void tdm_load_2d(unsigned lds_byte_addr,
                                            const void* gptr,
                                            unsigned tile_w, unsigned tile_h,
                                            unsigned tensor_w, unsigned tensor_h,
                                            unsigned stride) {
  const uint64_t ga = (uint64_t)(uintptr_t)gptr;
  v4u g0;
  g0[0] = 1u;                                   // count=1, user-mode descriptor
  g0[1] = lds_byte_addr;                        // LDS destination (bytes)
  g0[2] = (unsigned)ga;                         // global_addr[31:0]
  g0[3] = ((unsigned)(ga >> 32) & 0x01FFFFFFu) | 0x80000000u;  // addr[56:32]|type=2
  v8i g1;
  g1[0] = (int)(1u << 16);                      // data_size=1 -> 2 bytes/elem
  g1[1] = (int)((tensor_w & 0xFFFFu) << 16);                    // dim0[15:0]
  g1[2] = (int)((tensor_w >> 16) | ((tensor_h & 0xFFFFu) << 16));
  g1[3] = (int)((tensor_h >> 16) | (tile_w << 16));             // tile_dim0
  g1[4] = (int)tile_h;                          // tile_dim1 (tile_dim2 = 0)
  g1[5] = (int)stride;                          // tensor_dim0_stride[31:0]
  g1[6] = 0;                                    // stride hi | dim1_stride lo
  g1[7] = 0;
  const v4i z4 = {0, 0, 0, 0};
#if __clang_major__ >= 23
  const v8i z8 = {0, 0, 0, 0, 0, 0, 0, 0};
  __builtin_amdgcn_tensor_load_to_lds(g0, g1, z4, z4, z8, 0);
#else
  __builtin_amdgcn_tensor_load_to_lds(g0, g1, z4, z4, 0);
#endif
}

// ---------------- kernel 0: f32 -> bf16 ----------------
__global__ void cvt_f32_bf16(const float* __restrict__ in, __bf16* __restrict__ out,
                             int nelem) {
  int i = blockIdx.x * blockDim.x + threadIdx.x;
  int stride = gridDim.x * blockDim.x;
  for (; i < nelem; i += stride) out[i] = (__bf16)in[i];
}

// ---------------- kernel 1: QKV projection ----------------
// Block = 8 waves computes a 128x64 macro-tile of C = Xb @ Wb^T + bias.
// A(128x32) and B(64x32) panels are TDM double-buffered in LDS.
__global__ void qkv_gemm(const __bf16* __restrict__ xb, const __bf16* __restrict__ wb,
                         const float* __restrict__ bias,
                         __bf16* __restrict__ q, __bf16* __restrict__ k,
                         __bf16* __restrict__ vt) {
  __shared__ alignas(16) __bf16 sA[2][128 * 32];   // 16 KiB
  __shared__ alignas(16) __bf16 sB[2][64 * 32];    //  8 KiB
  const int lane = threadIdx.x & 31, wave = threadIdx.x >> 5;
  const int n = lane & 15, hi = lane >> 4;
  const int tM0 = (blockIdx.x / (NQKV / 64)) * 128;
  const int tN0 = (blockIdx.x % (NQKV / 64)) * 64;
  constexpr int KSTEPS = DIMN / 32;

  if (wave == 0) {
    tdm_load_2d(lds_off(&sA[0][0]), xb + (size_t)tM0 * DIMN, 32, 128, DIMN, MROWS, DIMN);
    tdm_load_2d(lds_off(&sB[0][0]), wb + (size_t)tN0 * DIMN, 32, 64,  DIMN, NQKV,  DIMN);
  }

  v8f acc[4] = {};
  int buf = 0;
  for (int i = 0; i < KSTEPS; ++i) {
    __syncthreads();  // prior compute done -> buf^1 is free to overwrite
    if (wave == 0) {
      if (i + 1 < KSTEPS) {
        const int k0 = (i + 1) * 32;
        tdm_load_2d(lds_off(&sA[buf ^ 1][0]), xb + (size_t)tM0 * DIMN + k0,
                    32, 128, DIMN, MROWS, DIMN);
        tdm_load_2d(lds_off(&sB[buf ^ 1][0]), wb + (size_t)tN0 * DIMN + k0,
                    32, 64, DIMN, NQKV, DIMN);
        __builtin_amdgcn_s_wait_tensorcnt(2);   // buf's 2 TDMs done; next pair flies
      } else {
        __builtin_amdgcn_s_wait_tensorcnt(0);
      }
    }
    __syncthreads();  // buf ready for everyone
    v16bf a = frag16(&sA[buf][(wave * 16 + n) * 32], hi);
#pragma unroll
    for (int t = 0; t < 4; ++t)
      acc[t] = wmma_bf16(a, frag16(&sB[buf][(t * 16 + n) * 32], hi), acc[t]);
    buf ^= 1;
  }

  // Epilogue: + bias, scatter into Q(scaled)/K row-major and V transposed.
#pragma unroll
  for (int t = 0; t < 4; ++t) {
    const int cg = tN0 + t * 16 + n;     // output column in [0,3072)
    const float bv = bias[cg];
    const int which = cg / DIMN;         // 0=q 1=k 2=v
    const int hh = (cg % DIMN) / HD;
    const int d  = cg % HD;
#pragma unroll
    for (int v = 0; v < 8; ++v) {
      const int r  = tM0 + wave * 16 + v + 8 * hi;
      const int b_ = r / LSEQ, l = r % LSEQ;
      const float val = acc[t][v] + bv;
      const size_t bh = (size_t)(b_ * NHEAD + hh);
      if (which == 0)      q [bh * LSEQ * HD + (size_t)l * HD + d] = (__bf16)(val * SCALE);
      else if (which == 1) k [bh * LSEQ * HD + (size_t)l * HD + d] = (__bf16)val;
      else                 vt[bh * HD * LSEQ + (size_t)d * LSEQ + l] = (__bf16)val;
    }
  }
}

// ---------------- kernel 2: flash attention ----------------
// Block = 4 waves = 64 query rows of one (b,h). K/V tiles are TDM
// double-buffered in LDS and shared by all 4 waves (4x traffic reduction).
__global__ void attn_kernel(const __bf16* __restrict__ q, const __bf16* __restrict__ kmat,
                            const __bf16* __restrict__ vt, __bf16* __restrict__ ob) {
  __shared__ alignas(16) __bf16 sK[2][32 * 64];    // 8 KiB: K rows j0..j0+31
  __shared__ alignas(16) __bf16 sV[2][64 * 32];    // 8 KiB: V^T rows d=0..63
  __shared__ alignas(16) __bf16 ldsP[4][16][32];   // 4 KiB: per-wave P transpose
  const int lane = threadIdx.x & 31;
  const int wave = threadIdx.x >> 5;
  const int bh   = blockIdx.x >> 5;
  const int qblk = blockIdx.x & 31;
  const int q0   = qblk * 64 + wave * 16;
  const int n = lane & 15, hi = lane >> 4;

  const __bf16* qb = q    + (size_t)bh * LSEQ * HD;
  const __bf16* kb = kmat + (size_t)bh * LSEQ * HD;
  const __bf16* vb = vt   + (size_t)bh * HD * LSEQ;

  const v16bf qa0 = gfrag(qb, HD, q0 + n, 0, hi);
  const v16bf qa1 = gfrag(qb, HD, q0 + n, 32, hi);

  v8f o0 = {}, o1 = {}, o2 = {}, o3 = {};
  float mrow[8], lrow[8];
#pragma unroll
  for (int v = 0; v < 8; ++v) { mrow[v] = -1e30f; lrow[v] = 0.0f; }

  if (wave == 0) {
    tdm_load_2d(lds_off(&sK[0][0]), kb, 64, 32, HD, LSEQ, HD);
    tdm_load_2d(lds_off(&sV[0][0]), vb, 32, 64, LSEQ, HD, LSEQ);
  }

  constexpr int JSTEPS = LSEQ / 32;
  int buf = 0;
  for (int i = 0; i < JSTEPS; ++i) {
    __syncthreads();
    if (wave == 0) {
      if (i + 1 < JSTEPS) {
        const int j1 = (i + 1) * 32;
        tdm_load_2d(lds_off(&sK[buf ^ 1][0]), kb + (size_t)j1 * HD, 64, 32, HD, LSEQ, HD);
        tdm_load_2d(lds_off(&sV[buf ^ 1][0]), vb + j1, 32, 64, LSEQ, HD, LSEQ);
        __builtin_amdgcn_s_wait_tensorcnt(2);
      } else {
        __builtin_amdgcn_s_wait_tensorcnt(0);
      }
    }
    __syncthreads();

    // S = Q @ K^T (Q pre-scaled): two 16x16 tiles, K-dim = 64, from LDS
    v8f s0 = {}, s1 = {};
    s0 = wmma_bf16(qa0, frag16(&sK[buf][(n)*64 + 0],  hi), s0);
    s0 = wmma_bf16(qa1, frag16(&sK[buf][(n)*64 + 32], hi), s0);
    s1 = wmma_bf16(qa0, frag16(&sK[buf][(16 + n)*64 + 0],  hi), s1);
    s1 = wmma_bf16(qa1, frag16(&sK[buf][(16 + n)*64 + 32], hi), s1);

    // Online softmax; row m = v + 8*hi lives across a 16-lane half-group.
#pragma unroll
    for (int v = 0; v < 8; ++v) {
      float mx = fmaxf(s0[v], s1[v]);
      mx = fmaxf(mx, __shfl_xor(mx, 1, 32));
      mx = fmaxf(mx, __shfl_xor(mx, 2, 32));
      mx = fmaxf(mx, __shfl_xor(mx, 4, 32));
      mx = fmaxf(mx, __shfl_xor(mx, 8, 32));
      const float mnew  = fmaxf(mrow[v], mx);
      const float alpha = __expf(mrow[v] - mnew);
      const float p0 = __expf(s0[v] - mnew);
      const float p1 = __expf(s1[v] - mnew);
      float rs = p0 + p1;
      rs += __shfl_xor(rs, 1, 32);
      rs += __shfl_xor(rs, 2, 32);
      rs += __shfl_xor(rs, 4, 32);
      rs += __shfl_xor(rs, 8, 32);
      lrow[v] = lrow[v] * alpha + rs;
      mrow[v] = mnew;
      o0[v] *= alpha; o1[v] *= alpha; o2[v] *= alpha; o3[v] *= alpha;
      const int m = v + 8 * hi;
      ldsP[wave][m][n]      = (__bf16)p0;
      ldsP[wave][m][16 + n] = (__bf16)p1;
    }
    // Per-wave staging region: DS ops are in-order per wave, dscnt drain only.
    asm volatile("s_wait_dscnt 0" ::: "memory");
    const v16bf pa = frag16(&ldsP[wave][n][0], hi);

    // O += P @ V from the shared V tile.
    o0 = wmma_bf16(pa, frag16(&sV[buf][(0  + n) * 32], hi), o0);
    o1 = wmma_bf16(pa, frag16(&sV[buf][(16 + n) * 32], hi), o1);
    o2 = wmma_bf16(pa, frag16(&sV[buf][(32 + n) * 32], hi), o2);
    o3 = wmma_bf16(pa, frag16(&sV[buf][(48 + n) * 32], hi), o3);
    buf ^= 1;
  }

  const int b_ = bh >> 4, h = bh & 15;
#pragma unroll
  for (int v = 0; v < 8; ++v) {
    const float inv = 1.0f / lrow[v];
    const int m = v + 8 * hi;
    const size_t row = (size_t)(b_ * LSEQ + q0 + m) * DIMN + h * HD;
    ob[row +  0 + n] = (__bf16)(o0[v] * inv);
    ob[row + 16 + n] = (__bf16)(o1[v] * inv);
    ob[row + 32 + n] = (__bf16)(o2[v] * inv);
    ob[row + 48 + n] = (__bf16)(o3[v] * inv);
  }
}

// ---------------- kernel 3: output projection ----------------
__global__ void out_gemm(const __bf16* __restrict__ ab, const __bf16* __restrict__ wb,
                         const float* __restrict__ bias, float* __restrict__ out) {
  __shared__ alignas(16) __bf16 sA[2][128 * 32];
  __shared__ alignas(16) __bf16 sB[2][64 * 32];
  const int lane = threadIdx.x & 31, wave = threadIdx.x >> 5;
  const int n = lane & 15, hi = lane >> 4;
  const int tM0 = (blockIdx.x / (DIMN / 64)) * 128;
  const int tN0 = (blockIdx.x % (DIMN / 64)) * 64;
  constexpr int KSTEPS = DIMN / 32;

  if (wave == 0) {
    tdm_load_2d(lds_off(&sA[0][0]), ab + (size_t)tM0 * DIMN, 32, 128, DIMN, MROWS, DIMN);
    tdm_load_2d(lds_off(&sB[0][0]), wb + (size_t)tN0 * DIMN, 32, 64,  DIMN, DIMN,  DIMN);
  }
  v8f acc[4] = {};
  int buf = 0;
  for (int i = 0; i < KSTEPS; ++i) {
    __syncthreads();
    if (wave == 0) {
      if (i + 1 < KSTEPS) {
        const int k0 = (i + 1) * 32;
        tdm_load_2d(lds_off(&sA[buf ^ 1][0]), ab + (size_t)tM0 * DIMN + k0,
                    32, 128, DIMN, MROWS, DIMN);
        tdm_load_2d(lds_off(&sB[buf ^ 1][0]), wb + (size_t)tN0 * DIMN + k0,
                    32, 64, DIMN, DIMN, DIMN);
        __builtin_amdgcn_s_wait_tensorcnt(2);
      } else {
        __builtin_amdgcn_s_wait_tensorcnt(0);
      }
    }
    __syncthreads();
    v16bf a = frag16(&sA[buf][(wave * 16 + n) * 32], hi);
#pragma unroll
    for (int t = 0; t < 4; ++t)
      acc[t] = wmma_bf16(a, frag16(&sB[buf][(t * 16 + n) * 32], hi), acc[t]);
    buf ^= 1;
  }
#pragma unroll
  for (int t = 0; t < 4; ++t) {
    const int cg = tN0 + t * 16 + n;
    const float bv = bias[cg];
#pragma unroll
    for (int v = 0; v < 8; ++v) {
      const int r = tM0 + wave * 16 + v + 8 * hi;
      out[(size_t)r * DIMN + cg] = acc[t][v] + bv;
    }
  }
}

// ---------------- host launcher ----------------
extern "C" void kernel_launch(void* const* d_in, const int* in_sizes, int n_in,
                              void* d_out, int out_size, void* d_ws, size_t ws_size,
                              hipStream_t stream) {
  const float* x     = (const float*)d_in[0];
  const float* qkv_w = (const float*)d_in[1];
  const float* qkv_b = (const float*)d_in[2];
  const float* out_w = (const float*)d_in[3];
  const float* out_b = (const float*)d_in[4];

  char* ws = (char*)d_ws;
  const size_t MB = 1024ull * 1024ull;
  __bf16* xb  = (__bf16*)(ws +  0 * MB);  // 8 MiB  x bf16
  __bf16* wqk = (__bf16*)(ws +  8 * MB);  // 6 MiB  qkv_w bf16
  __bf16* wo  = (__bf16*)(ws + 14 * MB);  // 2 MiB  out_w bf16
  __bf16* qf  = (__bf16*)(ws + 16 * MB);  // 8 MiB  Q (B,H,L,hd) * SCALE
  __bf16* kf  = (__bf16*)(ws + 24 * MB);  // 8 MiB  K (B,H,L,hd)
  __bf16* vtf = (__bf16*)(ws + 32 * MB);  // 8 MiB  V^T (B,H,hd,L)
  __bf16* ao  = (__bf16*)(ws + 40 * MB);  // 8 MiB  attn out (B,L,DIM)

  cvt_f32_bf16<<<1024, 256, 0, stream>>>(x,     xb,  MROWS * DIMN);
  cvt_f32_bf16<<<1024, 256, 0, stream>>>(qkv_w, wqk, NQKV * DIMN);
  cvt_f32_bf16<<< 512, 256, 0, stream>>>(out_w, wo,  DIMN * DIMN);

  qkv_gemm<<<(MROWS / 128) * (NQKV / 64), 256, 0, stream>>>(xb, wqk, qkv_b,
                                                            qf, kf, vtf);
  attn_kernel<<<BATCH * NHEAD * (LSEQ / 64), 128, 0, stream>>>(qf, kf, vtf, ao);
  out_gemm<<<(MROWS / 128) * (DIMN / 64), 256, 0, stream>>>(ao, wo, out_b,
                                                            (float*)d_out);
  (void)in_sizes; (void)n_in; (void)out_size; (void)ws_size;
}
